// LookupMemoryModule_61976378081396
// MI455X (gfx1250) — matrix operand order repaired and checked by reading, not compiled, not run on hardware
//
#include <hip/hip_runtime.h>
#include <float.h>

typedef __bf16 v16bf __attribute__((ext_vector_type(16)));
typedef __bf16 v8bf  __attribute__((ext_vector_type(8)));
typedef float  v8f   __attribute__((ext_vector_type(8)));
typedef unsigned int u32x4 __attribute__((ext_vector_type(4)));
typedef int          i32x8 __attribute__((ext_vector_type(8)));
typedef int          i32x4 __attribute__((ext_vector_type(4)));

#define ROWS_TOTAL 4096   // B*T
#define DIM        1024
#define NKEYS      65536
#define TOPKK      32

// ---------------- helpers ----------------

__device__ __forceinline__ unsigned short f2bf(float f) {
  unsigned u = __float_as_uint(f);
  u += 0x7FFFu + ((u >> 16) & 1u);        // round-to-nearest-even
  return (unsigned short)(u >> 16);
}

__global__ void cvt_f32_bf16(const float* __restrict__ in,
                             unsigned short* __restrict__ out, int n) {
  for (int i = blockIdx.x * blockDim.x + threadIdx.x; i < n;
       i += gridDim.x * blockDim.x)
    out[i] = f2bf(in[i]);
}

// TDM: DMA a 16-row x 2048-byte (32 KB) bf16 tile, global -> LDS.
// D# per CDNA5 ISA ch.8: group0 = {count=1 | lds_addr | global_addr | type=2},
// group1 = {data_size=8B, tensor 256x16 (8B units), tile 256x16, stride 256}.
// Issued by one wave (EXEC-independent, per-wave DMA); tracked by TENSORcnt.
// This toolchain exposes the 6-arg builtin: (g0, g1, g2, g3, g_extra, cpol).
__device__ __forceinline__ void tdm_load_tile_32k(unsigned lds_off,
                                                  const void* gsrc) {
  const unsigned long long ga = (unsigned long long)gsrc;
  u32x4 g0;
  g0.x = 1u;                                               // count=1 (valid), user
  g0.y = lds_off;                                          // lds_addr (bytes)
  g0.z = (unsigned)(ga & 0xffffffffu);                     // global_addr[31:0]
  g0.w = (unsigned)((ga >> 32) & 0x1ffffffu) | (2u << 30); // addr[56:32] | type=2
  i32x8 g1;
  g1[0] = 0x00030000;                                      // data_size = 8 bytes
  g1[1] = (int)(256u << 16);                               // tensor_dim0 = 256
  g1[2] = (int)(16u  << 16);                               // tensor_dim1 = 16
  g1[3] = (int)(256u << 16);                               // tile_dim0   = 256
  g1[4] = 16;                                              // tile_dim1   = 16
  g1[5] = 256;                                             // tensor_dim0_stride
  g1[6] = 0;
  g1[7] = 0;
  const i32x4 z4 = {0, 0, 0, 0};
  const i32x8 z8 = {0, 0, 0, 0, 0, 0, 0, 0};
  __builtin_amdgcn_tensor_load_to_lds(g0, g1, z4, z4, z8, 0);
}

// D = A(16xDIM, LDS) x B(colsxDIM rows, global)^T for one 16x16 tile.
// A-frag layout (16-bit A 16x32): lane<16 holds K=[k0..k0+8),[k0+16..k0+24);
// lane>=16 holds K=[k0+8..k0+16),[k0+24..k0+32). M = lane&15.
// B-frag layout: lane<16 -> col=lane, K=[k0..k0+8),[k0+16..k0+24); lane>=16 -> col=lane-16,
// K=[k0+8..k0+16),[k0+24..k0+32) (same K order as the A-frag).
__device__ __forceinline__ v8f wmma_rowtile(const unsigned short* sh_a,
                                            const unsigned short* __restrict__ bmat,
                                            int col_base, int lane, v8f acc) {
  const int hi = (lane >= 16);
  const int m  = lane & 15;
  const unsigned short* brow = bmat + (size_t)(col_base + m) * DIM + (hi ? 8 : 0);
  const unsigned short* arow = sh_a + m * DIM + (hi ? 8 : 0);
#pragma unroll 4
  for (int k0 = 0; k0 < DIM; k0 += 32) {
    v16bf a;
    *((v8bf*)&a)       = *(const v8bf*)(arow + k0);
    *(((v8bf*)&a) + 1) = *(const v8bf*)(arow + k0 + 16);
    v16bf b;
    *((v8bf*)&b)       = *(const v8bf*)(brow + k0);
    *(((v8bf*)&b) + 1) = *(const v8bf*)(brow + k0 + 16);
    acc = __builtin_amdgcn_wmma_f32_16x16x32_bf16(false, a, false, b,
                                                  (short)0, acc, false, false);
  }
  return acc;
}

// ---------------- q projection: q = x @ q_w^T (bf16 in, bf16 out) ----------------

__global__ void __launch_bounds__(256)
qproj_wmma(const unsigned short* __restrict__ xb,
           const unsigned short* __restrict__ wb,
           unsigned short* __restrict__ qb) {
  __shared__ __align__(16) unsigned short sh_a[16 * DIM];
  const int tid      = threadIdx.x;
  const int row_base = blockIdx.y * 16;
  const int col_tile = blockIdx.x * 128;
  if ((tid >> 5) == 0) {   // wave 0 issues the tile DMA
    tdm_load_tile_32k((unsigned)(size_t)&sh_a[0],
                      xb + (size_t)row_base * DIM);
    __builtin_amdgcn_s_wait_tensorcnt(0);
  }
  __syncthreads();
  const int wave = tid >> 5, lane = tid & 31;
  v8f acc = {};
  acc = wmma_rowtile(sh_a, wb, col_tile + wave * 16, lane, acc);
  // C layout: VGPR v -> M = v + (lane<16 ? 0 : 8), N = lane&15
  const int c  = col_tile + wave * 16 + (lane & 15);
  const int r0 = row_base + ((lane < 16) ? 0 : 8);
#pragma unroll
  for (int v = 0; v < 8; ++v)
    qb[(size_t)(r0 + v) * DIM + c] = f2bf(acc[v]);
}

// ---------------- fused scores GEMM + streaming top-32 ----------------

struct ScoresSmem {
  unsigned short sh_q[16 * DIM];   // 32 KB A tile (TDM destination: keep first)
  float sh_cand[16 * 128];
  float sh_tv[16 * TOPKK];
  int   sh_ti[16 * TOPKK];
  float sh_min[16];
};

__global__ void __launch_bounds__(256)
scores_topk_wmma(const unsigned short* __restrict__ qb,
                 const unsigned short* __restrict__ kb,
                 float* __restrict__ topv, int* __restrict__ topi) {
  __shared__ __align__(16) ScoresSmem sm;

  const int tid      = threadIdx.x;
  const int row_base = blockIdx.x * 16;
  if ((tid >> 5) == 0) {   // wave 0: DMA the 16x1024 bf16 q tile into LDS
    tdm_load_tile_32k((unsigned)(size_t)&sm.sh_q[0],
                      qb + (size_t)row_base * DIM);
    __builtin_amdgcn_s_wait_tensorcnt(0);
  }
  for (int i = tid; i < 16 * TOPKK; i += 256) { sm.sh_tv[i] = -FLT_MAX; sm.sh_ti[i] = 0; }
  if (tid < 16) sm.sh_min[tid] = -FLT_MAX;
  __syncthreads();

  const int wave = tid >> 5, lane = tid & 31;
  const int hi = lane >= 16;

  for (int nb = 0; nb < NKEYS; nb += 128) {
    if (nb + 128 < NKEYS)   // warm L2/WGP$ for the next key tile
      __builtin_prefetch(kb + (size_t)(nb + 128 + wave * 16 + (lane & 15)) * DIM, 0, 1);

    v8f acc = {};
    acc = wmma_rowtile(sm.sh_q, kb, nb + wave * 16, lane, acc);

    const int c  = wave * 16 + (lane & 15);
    const int r0 = hi ? 8 : 0;
#pragma unroll
    for (int v = 0; v < 8; ++v)
      sm.sh_cand[(r0 + v) * 128 + c] = acc[v];
    __syncthreads();

    // wave w maintains rows 2w, 2w+1: ballot-filter 32 candidates at a time,
    // rare survivors take the serial 32-entry replace path.
#pragma unroll
    for (int rr = 0; rr < 2; ++rr) {
      const int r = wave * 2 + rr;
      float* tv = &sm.sh_tv[r * TOPKK];
      int*   ti = &sm.sh_ti[r * TOPKK];
      for (int chunk = 0; chunk < 4; ++chunk) {
        const float val = sm.sh_cand[r * 128 + chunk * 32 + lane];
        unsigned long long mask = __ballot(val > sm.sh_min[r]);
        while (mask) {
          const int src = (int)__builtin_ctzll(mask);
          mask &= mask - 1;
          const float v2 = __shfl(val, src);
          const int   c2 = nb + chunk * 32 + src;
          if (lane == 0) {
            int mp = 0; float mv = tv[0];
#pragma unroll
            for (int i = 1; i < TOPKK; ++i) {
              const float t = tv[i];
              if (t < mv) { mv = t; mp = i; }
            }
            if (v2 > mv) {
              tv[mp] = v2; ti[mp] = c2;
              float nm = tv[0];
#pragma unroll
              for (int i = 1; i < TOPKK; ++i) nm = fminf(nm, tv[i]);
              sm.sh_min[r] = nm;
            }
          }
        }
      }
      asm volatile("s_wait_dscnt 0x0" ::: "memory");  // list writes visible in-wave
    }
    __syncthreads();
  }

  for (int e = tid; e < 16 * TOPKK; e += 256) {
    const int r = e >> 5, k = e & 31;
    topv[(size_t)(row_base + r) * TOPKK + k] = sm.sh_tv[e];
    topi[(size_t)(row_base + r) * TOPKK + k] = sm.sh_ti[e];
  }
}

// ---------------- softmax over top-32 + value gather ----------------

__global__ void __launch_bounds__(256)
softmax_gather(const float* __restrict__ topv, const int* __restrict__ topi,
               const float* __restrict__ values, float* __restrict__ out) {
  __shared__ float sw[TOPKK];
  __shared__ int   si[TOPKK];
  const int row = blockIdx.x;
  const int tid = threadIdx.x;
  if (tid < TOPKK) {                       // exactly wave 0 on wave32
    float v  = topv[(size_t)row * TOPKK + tid];
    si[tid]  = topi[(size_t)row * TOPKK + tid];
    float mx = v;
#pragma unroll
    for (int o = 16; o > 0; o >>= 1) mx = fmaxf(mx, __shfl_xor(mx, o));
    const float e = __expf(v - mx);
    float s = e;
#pragma unroll
    for (int o = 16; o > 0; o >>= 1) s += __shfl_xor(s, o);
    sw[tid] = e / s;
  }
  __syncthreads();
  for (int d = tid; d < DIM; d += 256) {
    float acc = 0.f;
#pragma unroll 8
    for (int k = 0; k < TOPKK; ++k)
      acc += sw[k] * values[(size_t)si[k] * DIM + d];
    out[(size_t)row * DIM + d] = acc;
  }
}

// ---------------- launch ----------------

extern "C" void kernel_launch(void* const* d_in, const int* in_sizes, int n_in,
                              void* d_out, int out_size, void* d_ws, size_t ws_size,
                              hipStream_t stream) {
  const float* x      = (const float*)d_in[0];
  const float* q_w    = (const float*)d_in[1];
  const float* keys   = (const float*)d_in[2];
  const float* values = (const float*)d_in[3];
  float* out = (float*)d_out;

  char*  ws  = (char*)d_ws;
  size_t off = 0;
  auto alloc = [&](size_t bytes) {
    void* p = ws + off;
    off = (off + bytes + 255) & ~(size_t)255;
    return p;
  };
  unsigned short* xb   = (unsigned short*)alloc((size_t)ROWS_TOTAL * DIM * 2); //   8 MB
  unsigned short* wb   = (unsigned short*)alloc((size_t)DIM * DIM * 2);        //   2 MB
  unsigned short* kb   = (unsigned short*)alloc((size_t)NKEYS * DIM * 2);      // 128 MB (L2-resident)
  unsigned short* qb   = (unsigned short*)alloc((size_t)ROWS_TOTAL * DIM * 2); //   8 MB
  float*          topv = (float*)alloc((size_t)ROWS_TOTAL * TOPKK * 4);
  int*            topi = (int*)alloc((size_t)ROWS_TOTAL * TOPKK * 4);

  cvt_f32_bf16<<<2048, 256, 0, stream>>>(x,    xb, ROWS_TOTAL * DIM);
  cvt_f32_bf16<<<1024, 256, 0, stream>>>(q_w,  wb, DIM * DIM);
  cvt_f32_bf16<<<8192, 256, 0, stream>>>(keys, kb, NKEYS * DIM);
  qproj_wmma<<<dim3(DIM / 128, ROWS_TOTAL / 16), 256, 0, stream>>>(xb, wb, qb);
  scores_topk_wmma<<<ROWS_TOTAL / 16, 256, 0, stream>>>(qb, kb, topv, topi);
  softmax_gather<<<ROWS_TOTAL, 256, 0, stream>>>(topv, topi, values, out);
}
